// SynapticDegeneracy_12704513261981
// MI455X (gfx1250) — matrix operand order repaired and checked by reading, not compiled, not run on hardware
//
#include <hip/hip_runtime.h>
#include <hip/hip_bf16.h>

typedef __attribute__((ext_vector_type(16))) _Float16 v16h;
typedef __attribute__((ext_vector_type(8)))  float    v8f;

#define D_NODE 64
#define HIDDEN 64
#define CHUNK  256   // edges per compaction block

// ---------------------------------------------------------------------------
// Kernel 1: edge-drop probability MLP via WMMA, emit alive flags.
//   A tile = 16 edges x 128 features (f16), B = W1 (128x64, f16 in LDS),
//   C = f32 accum.  16 x v_wmma_f32_16x16x32_f16 per tile.
// ---------------------------------------------------------------------------
__global__ __launch_bounds__(256)
void sd_mlp_flags_kernel(const float* __restrict__ nodes,
                         const int*   __restrict__ rec,
                         const int*   __restrict__ send,
                         const float* __restrict__ aedges,
                         const float* __restrict__ u,
                         const float* __restrict__ W1,
                         const float* __restrict__ b1,
                         const float* __restrict__ W2,
                         const float* __restrict__ b2,
                         unsigned char* __restrict__ flags,
                         int num_tiles)
{
    __shared__ _Float16 w1h[2 * D_NODE * HIDDEN]; // 128x64 f16 = 16 KB
    __shared__ float    b1s[HIDDEN];
    __shared__ float    w2s[HIDDEN];

    // Stage W1 (f32 -> f16), b1, W2 into LDS once per block.
    for (int i = threadIdx.x; i < 2 * D_NODE * HIDDEN; i += blockDim.x)
        w1h[i] = (_Float16)W1[i];
    if (threadIdx.x < HIDDEN) {
        b1s[threadIdx.x] = b1[threadIdx.x];
        w2s[threadIdx.x] = W2[threadIdx.x];
    }
    __syncthreads();

    const int lane  = threadIdx.x & 31;
    const int nl    = lane & 15;   // column within 16 (B/C), row M for A
    const int khalf = lane >> 4;   // which K half this lane holds
    const float b2v = b2[0];

    // Hoist the 16 W1 fragments (invariant across tiles) into registers.
    // B 32x16 f16 layout: lane -> column nl, elems 0..7 -> K = kb*32 + khalf*8 + j,
    // elems 8..15 -> K = kb*32 + 16 + khalf*8 + (j-8).
    v16h Bf[16];
#pragma unroll
    for (int kb = 0; kb < 4; ++kb) {
#pragma unroll
        for (int t = 0; t < 4; ++t) {
            v16h bf;
#pragma unroll
            for (int j = 0; j < 16; ++j)
                bf[j] = w1h[(kb * 32 + (j < 8 ? khalf * 8 + j : 16 + khalf * 8 + (j - 8))) * HIDDEN + (t * 16 + nl)];
            Bf[kb * 4 + t] = bf;
        }
    }

    const int gwave  = (blockIdx.x * blockDim.x + threadIdx.x) >> 5;
    const int nwaves = (gridDim.x * blockDim.x) >> 5;

    for (int tile = gwave; tile < num_tiles; tile += nwaves) {
        const int e0 = tile * 16;
        const int m  = nl;                 // A row handled by this lane
        const int e  = e0 + m;
        const int r  = rec[e];
        const int s  = send[e];

        v8f acc[4];
#pragma unroll
        for (int t = 0; t < 4; ++t)
#pragma unroll
            for (int i = 0; i < 8; ++i) acc[t][i] = 0.0f;

        // A 16x32 f16 layout per lane: elems 0..7 -> K = kb*32 + khalf*8 + j,
        // elems 8..15 -> K = kb*32 + 16 + khalf*8 + j.
        // feat = [nodes[r] (K 0..63) | nodes[s] (K 64..127)], so each kb chunk
        // is 32 contiguous floats of one node row.
#pragma unroll
        for (int kb = 0; kb < 4; ++kb) {
            const float* src = (kb < 2)
                ? (nodes + (size_t)r * D_NODE + kb * 32)
                : (nodes + (size_t)s * D_NODE + (kb - 2) * 32);
            const int o1 = khalf * 8;
            const int o2 = 16 + khalf * 8;
            float4 f0 = *(const float4*)(src + o1);
            float4 f1 = *(const float4*)(src + o1 + 4);
            float4 f2 = *(const float4*)(src + o2);
            float4 f3 = *(const float4*)(src + o2 + 4);
            v16h a;
            a[0]  = (_Float16)f0.x; a[1]  = (_Float16)f0.y;
            a[2]  = (_Float16)f0.z; a[3]  = (_Float16)f0.w;
            a[4]  = (_Float16)f1.x; a[5]  = (_Float16)f1.y;
            a[6]  = (_Float16)f1.z; a[7]  = (_Float16)f1.w;
            a[8]  = (_Float16)f2.x; a[9]  = (_Float16)f2.y;
            a[10] = (_Float16)f2.z; a[11] = (_Float16)f2.w;
            a[12] = (_Float16)f3.x; a[13] = (_Float16)f3.y;
            a[14] = (_Float16)f3.z; a[15] = (_Float16)f3.w;
#pragma unroll
            for (int t = 0; t < 4; ++t)
                acc[t] = __builtin_amdgcn_wmma_f32_16x16x32_f16(
                    false, a, false, Bf[kb * 4 + t],
                    (short)0, acc[t], false, false);
        }

        // Layer 2: per-lane partial of relu(C + b1) . W2, then butterfly
        // reduce over the 16 lanes of each half (C layout: lane = N column,
        // vgpr i = row M = i + 8*khalf).
        float p[8];
#pragma unroll
        for (int i = 0; i < 8; ++i) p[i] = 0.0f;
#pragma unroll
        for (int t = 0; t < 4; ++t) {
            const float b1v = b1s[t * 16 + nl];
            const float w2v = w2s[t * 16 + nl];
#pragma unroll
            for (int i = 0; i < 8; ++i) {
                float hv = acc[t][i] + b1v;
                hv = hv > 0.0f ? hv : 0.0f;
                p[i] += hv * w2v;
            }
        }
#pragma unroll
        for (int mask = 1; mask <= 8; mask <<= 1)
#pragma unroll
            for (int i = 0; i < 8; ++i)
                p[i] += __shfl_xor(p[i], mask, 32);

        if (nl == 0) { // lane 0 -> rows 0..7, lane 16 -> rows 8..15
#pragma unroll
            for (int i = 0; i < 8; ++i) {
                const int row = i + 8 * khalf;
                const int ee  = e0 + row;
                const float ae = aedges[ee];
                const float x  = p[i] + b2v;
                const float prob = ae / (1.0f + __expf(-x));
                const bool dead  = u[ee] < prob;          // degens
                const float nae  = dead ? 0.0f : ae;      // naedges value
                flags[ee] = (nae != 0.0f) ? 1 : 0;        // alive
            }
        }
    }
}

// ---------------------------------------------------------------------------
// Kernel 2a: alive count per 256-edge chunk.
// ---------------------------------------------------------------------------
__global__ __launch_bounds__(CHUNK)
void sd_count_kernel(const unsigned char* __restrict__ flags,
                     int* __restrict__ counts)
{
    __shared__ int cnt;
    if (threadIdx.x == 0) cnt = 0;
    __syncthreads();
    const int e = blockIdx.x * CHUNK + threadIdx.x;
    atomicAdd(&cnt, (int)flags[e]);
    __syncthreads();
    if (threadIdx.x == 0) counts[blockIdx.x] = cnt;
}

// ---------------------------------------------------------------------------
// Kernel 2b: single-block exclusive scan over chunk counts.
// ---------------------------------------------------------------------------
__global__ __launch_bounds__(256)
void sd_scan_kernel(const int* __restrict__ counts,
                    int* __restrict__ offsets,
                    int* __restrict__ total,
                    int nchunk)
{
    __shared__ int sd[256];
    __shared__ int carry_s;
    const int t = threadIdx.x;
    if (t == 0) carry_s = 0;
    __syncthreads();

    for (int base = 0; base < nchunk; base += 256) {
        const int idx = base + t;
        const int v = (idx < nchunk) ? counts[idx] : 0;
        sd[t] = v;
        __syncthreads();
#pragma unroll
        for (int off = 1; off < 256; off <<= 1) {
            int tmp = (t >= off) ? sd[t - off] : 0;
            __syncthreads();
            sd[t] += tmp;
            __syncthreads();
        }
        const int incl = sd[t];
        const int excl = incl - v;
        const int c = carry_s;
        if (idx < nchunk) offsets[idx] = c + excl;
        const int blockTotal = sd[255];
        __syncthreads();
        if (t == 0) carry_s = c + blockTotal;
        __syncthreads();
    }
    if (t == 0) total[0] = carry_s;
}

// ---------------------------------------------------------------------------
// Kernel 3: stable-partition scatter of all four outputs.
// ---------------------------------------------------------------------------
__global__ __launch_bounds__(CHUNK)
void sd_scatter_kernel(const unsigned char* __restrict__ flags,
                       const int*   __restrict__ offsets,
                       const int*   __restrict__ total,
                       const int*   __restrict__ rec,
                       const int*   __restrict__ send,
                       const float* __restrict__ edges,
                       float* __restrict__ naedges_o,
                       float* __restrict__ nrec_o,
                       float* __restrict__ nsend_o,
                       float* __restrict__ newe_o,
                       int max_nodes_m1)
{
    __shared__ int sd[CHUNK];
    const int t = threadIdx.x;
    const int e = blockIdx.x * CHUNK + t;
    const int flag = (int)flags[e];

    sd[t] = flag;
    __syncthreads();
#pragma unroll
    for (int off = 1; off < CHUNK; off <<= 1) {
        int tmp = (t >= off) ? sd[t - off] : 0;
        __syncthreads();
        sd[t] += tmp;
        __syncthreads();
    }
    const int exclAlive = sd[t] - flag;                 // alive before me in chunk
    const int aliveG = offsets[blockIdx.x] + exclAlive; // global alive excl prefix
    const int deadG  = e - aliveG;                      // global dead excl prefix
    const int dst    = flag ? aliveG : (total[0] + deadG);

    naedges_o[dst] = flag ? 1.0f : 0.0f;
    nrec_o[dst]    = flag ? (float)rec[e]  : (float)max_nodes_m1;
    nsend_o[dst]   = flag ? (float)send[e] : (float)max_nodes_m1;

    const float4* srcv = (const float4*)(edges + (size_t)e * 32);
    float4*       dstv = (float4*)(newe_o + (size_t)dst * 32);
    const float4 z4 = make_float4(0.f, 0.f, 0.f, 0.f);
#pragma unroll
    for (int j = 0; j < 8; ++j)
        dstv[j] = flag ? srcv[j] : z4;
}

// ---------------------------------------------------------------------------
extern "C" void kernel_launch(void* const* d_in, const int* in_sizes, int n_in,
                              void* d_out, int out_size, void* d_ws, size_t ws_size,
                              hipStream_t stream) {
    const float* nodes  = (const float*)d_in[0];
    const float* edges  = (const float*)d_in[1];
    const int*   rec    = (const int*)  d_in[2];
    const int*   send   = (const int*)  d_in[3];
    const float* aedges = (const float*)d_in[4];
    const float* u      = (const float*)d_in[5];
    const float* W1     = (const float*)d_in[6];
    const float* b1     = (const float*)d_in[7];
    const float* W2     = (const float*)d_in[8];
    const float* b2     = (const float*)d_in[9];

    const int E = in_sizes[2];              // 1,600,000
    const int NT = E / 16;                  // 100,000 WMMA tiles
    const int NCHUNK = E / CHUNK;           // 6,250

    // workspace layout
    unsigned char* flags = (unsigned char*)d_ws;                // E bytes
    int* counts  = (int*)((char*)d_ws + E);                     // NCHUNK
    int* offsets = counts + NCHUNK;                             // NCHUNK
    int* total   = offsets + NCHUNK;                            // 1

    // output layout (flattened tuple, f32)
    float* out       = (float*)d_out;
    float* naedges_o = out;
    float* nrec_o    = out + (size_t)E;
    float* nsend_o   = out + (size_t)2 * E;
    float* newe_o    = out + (size_t)3 * E;

    sd_mlp_flags_kernel<<<1024, 256, 0, stream>>>(
        nodes, rec, send, aedges, u, W1, b1, W2, b2, flags, NT);

    sd_count_kernel<<<NCHUNK, CHUNK, 0, stream>>>(flags, counts);

    sd_scan_kernel<<<1, 256, 0, stream>>>(counts, offsets, total, NCHUNK);

    sd_scatter_kernel<<<NCHUNK, CHUNK, 0, stream>>>(
        flags, offsets, total, rec, send, edges,
        naedges_o, nrec_o, nsend_o, newe_o, 50000 - 1);
}
